// Attention_11879879543330
// MI455X (gfx1250) — matrix operand order repaired and checked by reading, not compiled, not run on hardware
//
#include <hip/hip_runtime.h>
#include <hip/hip_bf16.h>

// ---------- problem constants ----------
#define BATCH 2
#define SEQ   2048
#define DMODEL 1024
#define NHEADS 16
#define HDIM  64
#define WINDOW 256
#define MTOT  (BATCH * SEQ)     // 4096
#define PADCOLS 288             // 272-col window padded to 9*32

typedef __attribute__((ext_vector_type(16))) __bf16 v16bf;
typedef __attribute__((ext_vector_type(8)))  __bf16 v8bf;
typedef __attribute__((ext_vector_type(8)))  float  v8f;
typedef __attribute__((ext_vector_type(4)))  unsigned int u32x4;
typedef __attribute__((ext_vector_type(4)))  int i32x4;
typedef __attribute__((ext_vector_type(8)))  int i32x8;

__device__ __forceinline__ __bf16 f2bf(float f) {
    unsigned u = __builtin_bit_cast(unsigned, f);
    u += 0x7FFFu + ((u >> 16) & 1u);           // round-to-nearest-even
    unsigned short h = (unsigned short)(u >> 16);
    return __builtin_bit_cast(__bf16, h);
}

__device__ __forceinline__ v16bf make_frag(const __bf16* p0, const __bf16* p1) {
    v8bf lo = *(const v8bf*)p0;
    v8bf hi = *(const v8bf*)p1;
    return __builtin_shufflevector(lo, hi, 0,1,2,3,4,5,6,7,8,9,10,11,12,13,14,15);
}

__device__ __forceinline__ v8f wmma_bf16(v16bf a, v16bf b, v8f c) {
    return __builtin_amdgcn_wmma_f32_16x16x32_bf16(
        false, a, false, b, (short)0, c, false, false);
}

// ---------- TDM: 2-D tile (tile_d0 x tile_d1 elements, bf16) global -> LDS ----
// D# per CDNA5 ISA. Tiles are always fully
// in-bounds, so tensor dims == tile dims; row pitch = stride0 elements.
__device__ __forceinline__ void tdm_load_2d(unsigned lds_off, const void* gptr,
                                            unsigned tile_d0, unsigned tile_d1,
                                            unsigned stride0_elems) {
    unsigned long long ga = (unsigned long long)(size_t)gptr;
    u32x4 g0;
    g0[0] = 1u;                                   // count=1, user descriptor
    g0[1] = lds_off;                              // lds_addr (bytes)
    g0[2] = (unsigned)(ga & 0xFFFFFFFFu);         // global_addr[31:0]
    g0[3] = (unsigned)((ga >> 32) & 0x01FFFFFFu)  // global_addr[56:32]
          | (2u << 30);                           // type = 2 ("image")

    i32x8 g1;
    g1[0] = (int)(1u << 16);                      // data_size = 2 bytes
    g1[1] = (int)((tile_d0 & 0xFFFFu) << 16);     // tensor_dim0[15:0]
    g1[2] = (int)(((tile_d0 >> 16) & 0xFFFFu)     // tensor_dim0[31:16]
          | ((tile_d1 & 0xFFFFu) << 16));         // tensor_dim1[15:0]
    g1[3] = (int)(((tile_d1 >> 16) & 0xFFFFu)     // tensor_dim1[31:16]
          | ((tile_d0 & 0xFFFFu) << 16));         // tile_dim0
    g1[4] = (int)(tile_d1 & 0xFFFFu);             // tile_dim1 (tile_dim2 = 0)
    g1[5] = (int)stride0_elems;                   // tensor_dim0_stride[31:0]
    g1[6] = 0;                                    // stride0[47:32], dim1_stride lo
    g1[7] = 0;

    i32x4 gz = {0, 0, 0, 0};
#if __has_builtin(__builtin_amdgcn_tensor_load_to_lds)
#if __clang_major__ >= 23
    i32x8 gz8 = {0, 0, 0, 0, 0, 0, 0, 0};
    __builtin_amdgcn_tensor_load_to_lds(g0, g1, gz, gz, gz8, 0);
#else
    __builtin_amdgcn_tensor_load_to_lds(g0, g1, gz, gz, 0);
#endif
#endif
}

__device__ __forceinline__ unsigned lds_off_of(const void* p) {
    return (unsigned)(size_t)p;                   // low 32 bits = LDS byte addr
}

// ---------- fp32 -> bf16 convert ----------
__global__ void cvt_bf16_kernel(const float* __restrict__ in,
                                __bf16* __restrict__ out, int n) {
    int i = blockIdx.x * blockDim.x + threadIdx.x;
    if (i < n) out[i] = f2bf(in[i]);
}

// ---------- GEMM: out[M,N] = A[M,K] @ W[N,K]^T  (bf16 in, f32 accum) --------
// TDM double-buffered LDS staging: per K-step, A tile 128x32 (8KB) and
// B tile 256x32 (16KB) DMA'd by wave 0, consumed by 8 waves (2x4, 64x64 each).
// MODE 0: bf16 row-major [M,N]; MODE 1: bf16 [b,h,hd,s]; MODE 2: f32 [M,N]
template <int MODE>
__global__ __launch_bounds__(256)
void gemm_bf16_kernel(const __bf16* __restrict__ A,
                      const __bf16* __restrict__ W,
                      void* __restrict__ outp,
                      int M, int N, int K) {
    __shared__ __align__(16) __bf16 sA[2][128 * 32];   // 2 x 8 KB
    __shared__ __align__(16) __bf16 sB[2][256 * 32];   // 2 x 16 KB

    const int lane  = threadIdx.x & 31;
    const int wave  = threadIdx.x >> 5;
    const int half  = lane >> 4;
    const int l15   = lane & 15;
    const int wm    = wave & 1;          // 2 waves along M
    const int wn    = wave >> 1;         // 4 waves along N
    const int mblk  = blockIdx.x * 128;
    const int nblk  = blockIdx.y * 256;

    const __bf16* Ablk = A + (size_t)mblk * K;
    const __bf16* Wblk = W + (size_t)nblk * K;

    // prologue: DMA first K-slab into buffer 0
    if (wave == 0) {
        tdm_load_2d(lds_off_of(&sA[0][0]), Ablk, 32, 128, (unsigned)K);
        tdm_load_2d(lds_off_of(&sB[0][0]), Wblk, 32, 256, (unsigned)K);
        __builtin_amdgcn_s_wait_tensorcnt(0);
    }
    __syncthreads();

    v8f zero = {0.f,0.f,0.f,0.f,0.f,0.f,0.f,0.f};
    v8f acc[4][4];
#pragma unroll
    for (int i = 0; i < 4; ++i)
#pragma unroll
        for (int j = 0; j < 4; ++j) acc[i][j] = zero;

    for (int kk = 0; kk < K; kk += 32) {
        const int buf = (kk >> 5) & 1;
        if (wave == 0 && kk + 32 < K) {          // prefetch next slab
            tdm_load_2d(lds_off_of(&sA[buf ^ 1][0]), Ablk + kk + 32, 32, 128, (unsigned)K);
            tdm_load_2d(lds_off_of(&sB[buf ^ 1][0]), Wblk + kk + 32, 32, 256, (unsigned)K);
        }

        v16bf af[4], bfr[4];
#pragma unroll
        for (int i = 0; i < 4; ++i) {
            const __bf16* ar = &sA[buf][(wm * 64 + i * 16 + l15) * 32];
            af[i] = make_frag(ar + half * 16, ar + half * 16 + 8);
        }
#pragma unroll
        for (int j = 0; j < 4; ++j) {
            const __bf16* br = &sB[buf][(wn * 64 + j * 16 + l15) * 32];
            bfr[j] = make_frag(br + half * 16, br + half * 16 + 8);
        }
#pragma unroll
        for (int i = 0; i < 4; ++i)
#pragma unroll
            for (int j = 0; j < 4; ++j)
                acc[i][j] = wmma_bf16(af[i], bfr[j], acc[i][j]);

        if (wave == 0) __builtin_amdgcn_s_wait_tensorcnt(0);
        __syncthreads();
    }

#pragma unroll
    for (int i = 0; i < 4; ++i) {
#pragma unroll
        for (int j = 0; j < 4; ++j) {
            int n = nblk + wn * 64 + j * 16 + l15;
#pragma unroll
            for (int r = 0; r < 8; ++r) {
                int m = mblk + wm * 64 + i * 16 + r + half * 8;
                float v = acc[i][j][r];
                if (MODE == 0) {
                    ((__bf16*)outp)[(size_t)m * N + n] = f2bf(v);
                } else if (MODE == 1) {
                    int b = m >> 11, s = m & (SEQ - 1);
                    int h = n >> 6, d = n & (HDIM - 1);
                    ((__bf16*)outp)[(size_t)((b * NHEADS + h) * HDIM + d) * SEQ + s] = f2bf(v);
                } else {
                    ((float*)outp)[(size_t)m * N + n] = v;
                }
            }
        }
    }
}

// ---------- sliding-window attention ----------
// one wave per 16-query tile; 2 waves / block
__global__ __launch_bounds__(64)
void attn_kernel(const __bf16* __restrict__ qb,
                 const __bf16* __restrict__ kb,
                 const __bf16* __restrict__ vt,   // [B, H, HDIM, SEQ]
                 __bf16* __restrict__ attnb) {    // [B*S, D]
    __shared__ __align__(16) float  s_sc[2][16][PADCOLS];
    __shared__ __align__(16) __bf16 s_p [2][16][PADCOLS];

    const int lane = threadIdx.x & 31;
    const int w    = threadIdx.x >> 5;
    const int half = lane >> 4;
    const int l15  = lane & 15;

    const int gw = blockIdx.x * 2 + w;
    const int qt = gw & 127;                  // 128 q-tiles per (b,h)
    const int h  = (gw >> 7) & (NHEADS - 1);
    const int b  = gw >> 11;
    const int q0 = qt * 16;
    const int j0 = q0 - 256;

    // Q fragments (A operand), d = 0..63 -> 2 frags of K=32
    const __bf16* qrow = qb + (size_t)(b * SEQ + q0 + l15) * DMODEL + h * HDIM;
    v16bf aq0 = make_frag(qrow + half * 16,      qrow + half * 16 + 8);
    v16bf aq1 = make_frag(qrow + 32 + half * 16, qrow + 32 + half * 16 + 8);

    // ---- scores: 17 key tiles of 16 ----
    for (int t = 0; t < 17; ++t) {
        const int j  = j0 + t * 16 + l15;                 // this lane's column
        const int jc = min(max(j, 0), SEQ - 1);
        const __bf16* krow = kb + (size_t)(b * SEQ + jc) * DMODEL + h * HDIM;
        v16bf b0 = make_frag(krow + half * 16,      krow + half * 16 + 8);
        v16bf b1 = make_frag(krow + 32 + half * 16, krow + 32 + half * 16 + 8);
        v8f acc = {0.f,0.f,0.f,0.f,0.f,0.f,0.f,0.f};
        acc = wmma_bf16(aq0, b0, acc);
        acc = wmma_bf16(aq1, b1, acc);
#pragma unroll
        for (int r = 0; r < 8; ++r) {
            int m = r + half * 8;
            int i = q0 + m;
            bool ok = (j >= 0) && (j <= i) && (j >= i - (WINDOW - 1));
            s_sc[w][m][t * 16 + l15] = ok ? acc[r] * 0.125f : -1.0e30f;
        }
    }
    // pad columns 272..287
#pragma unroll
    for (int r = 0; r < 8; ++r)
        s_sc[w][r + half * 8][272 + l15] = -1.0e30f;

    __syncthreads();

    // ---- softmax per row (wave-local, shfl reductions) ----
    for (int m = 0; m < 16; ++m) {
        float vals[9];
        float mx = -3.0e38f;
#pragma unroll
        for (int t = 0; t < 9; ++t) {
            float v = s_sc[w][m][lane + 32 * t];
            vals[t] = v;
            mx = fmaxf(mx, v);
        }
#pragma unroll
        for (int off = 16; off > 0; off >>= 1)
            mx = fmaxf(mx, __shfl_xor(mx, off, 32));
        float sum = 0.f;
#pragma unroll
        for (int t = 0; t < 9; ++t) {
            float e = __expf(vals[t] - mx);
            vals[t] = e;
            sum += e;
        }
#pragma unroll
        for (int off = 16; off > 0; off >>= 1)
            sum += __shfl_xor(sum, off, 32);
        float inv = 1.0f / sum;
#pragma unroll
        for (int t = 0; t < 9; ++t)
            s_p[w][m][lane + 32 * t] = f2bf(vals[t] * inv);
    }

    __syncthreads();

    // ---- out = P @ V : 4 d-tiles, 9 K-steps of 32 ----
    v8f oacc[4];
    v8f zero = {0.f,0.f,0.f,0.f,0.f,0.f,0.f,0.f};
#pragma unroll
    for (int dt = 0; dt < 4; ++dt) oacc[dt] = zero;

    const __bf16* prow = &s_p[w][l15][0];
    for (int k = 0; k < 9; ++k) {
        v16bf ap = make_frag(prow + k * 32 + half * 16,
                             prow + k * 32 + half * 16 + 8);
        int jc  = j0 + k * 32 + half * 16;
        int jcc = min(max(jc, 0), SEQ - 16);
#pragma unroll
        for (int dt = 0; dt < 4; ++dt) {
            int d = dt * 16 + l15;
            const __bf16* vrow = vt + (size_t)((b * NHEADS + h) * HDIM + d) * SEQ;
            v16bf bv = make_frag(vrow + jcc, vrow + jcc + 8);
            oacc[dt] = wmma_bf16(ap, bv, oacc[dt]);
        }
    }

#pragma unroll
    for (int dt = 0; dt < 4; ++dt) {
#pragma unroll
        for (int r = 0; r < 8; ++r) {
            int m = r + half * 8;
            attnb[(size_t)(b * SEQ + q0 + m) * DMODEL + h * HDIM + dt * 16 + l15] =
                f2bf(oacc[dt][r]);
        }
    }
}

// ---------- launch ----------
extern "C" void kernel_launch(void* const* d_in, const int* in_sizes, int n_in,
                              void* d_out, int out_size, void* d_ws, size_t ws_size,
                              hipStream_t stream) {
    (void)in_sizes; (void)n_in; (void)out_size; (void)ws_size;

    const float* x  = (const float*)d_in[0];
    const float* wq = (const float*)d_in[1];
    const float* wk = (const float*)d_in[2];
    const float* wv = (const float*)d_in[3];
    const float* wo = (const float*)d_in[4];
    float* out = (float*)d_out;

    const size_t XN = (size_t)MTOT * DMODEL;      // 4,194,304
    const size_t WN = (size_t)DMODEL * DMODEL;    // 1,048,576

    char* ws = (char*)d_ws;
    size_t off = 0;
    __bf16* xb   = (__bf16*)(ws + off); off += XN * 2;
    __bf16* wqb  = (__bf16*)(ws + off); off += WN * 2;
    __bf16* wkb  = (__bf16*)(ws + off); off += WN * 2;
    __bf16* wvb  = (__bf16*)(ws + off); off += WN * 2;
    __bf16* wob  = (__bf16*)(ws + off); off += WN * 2;
    __bf16* qb   = (__bf16*)(ws + off); off += XN * 2;
    __bf16* kb   = (__bf16*)(ws + off); off += XN * 2;
    __bf16* vtb  = (__bf16*)(ws + off); off += XN * 2;
    __bf16* attb = (__bf16*)(ws + off); off += XN * 2;

    // converts
    cvt_bf16_kernel<<<(int)(XN / 256), 256, 0, stream>>>(x,  xb,  (int)XN);
    cvt_bf16_kernel<<<(int)(WN / 256), 256, 0, stream>>>(wq, wqb, (int)WN);
    cvt_bf16_kernel<<<(int)(WN / 256), 256, 0, stream>>>(wk, wkb, (int)WN);
    cvt_bf16_kernel<<<(int)(WN / 256), 256, 0, stream>>>(wv, wvb, (int)WN);
    cvt_bf16_kernel<<<(int)(WN / 256), 256, 0, stream>>>(wo, wob, (int)WN);

    dim3 ggrid(MTOT / 128, DMODEL / 256);
    gemm_bf16_kernel<0><<<ggrid, 256, 0, stream>>>(xb, wqb, qb,  MTOT, DMODEL, DMODEL);
    gemm_bf16_kernel<0><<<ggrid, 256, 0, stream>>>(xb, wkb, kb,  MTOT, DMODEL, DMODEL);
    gemm_bf16_kernel<1><<<ggrid, 256, 0, stream>>>(xb, wvb, vtb, MTOT, DMODEL, DMODEL);

    // attention: one wave per 16-query tile, 2 waves per block
    const int nwaves = BATCH * NHEADS * (SEQ / 16);   // 4096
    attn_kernel<<<nwaves / 2, 64, 0, stream>>>(qb, kb, vtb, attb);

    gemm_bf16_kernel<2><<<ggrid, 256, 0, stream>>>(attb, wob, out, MTOT, DMODEL, DMODEL);
}
